// MultiHeadSelfAttention_57767310131586
// MI455X (gfx1250) — matrix operand order repaired and checked by reading, not compiled, not run on hardware
//
#include <hip/hip_runtime.h>

typedef __bf16 bf16;
typedef __attribute__((ext_vector_type(16))) __bf16 v16bf;
typedef __attribute__((ext_vector_type(8)))  float  v8f;
typedef __attribute__((ext_vector_type(4)))  unsigned int v4u;
typedef __attribute__((ext_vector_type(8)))  int v8i;
typedef __attribute__((ext_vector_type(4)))  int v4i;

typedef __attribute__((address_space(1))) void g1void;
typedef __attribute__((address_space(3))) void l3void;
typedef __attribute__((address_space(3))) char l3char;

#define WMMA_BF16(a, b, c) \
  __builtin_amdgcn_wmma_f32_16x16x32_bf16(false, (a), false, (b), (short)0, (c), false, false)

// ---- CDNA5 data-mover feature detection -----------------------------------
#if defined(__has_builtin)
#  if __has_builtin(__builtin_amdgcn_tensor_load_to_lds)
#    define USE_TDM 1
#  else
#    define USE_TDM 0
#  endif
#  if __has_builtin(__builtin_amdgcn_global_load_async_to_lds_b128)
#    define USE_ASYNC_LDS 1
#  else
#    define USE_ASYNC_LDS 0
#  endif
#  if __has_builtin(__builtin_amdgcn_s_wait_asynccnt)
#    define WAIT_ASYNCCNT() __builtin_amdgcn_s_wait_asynccnt(0)
#  else
#    define WAIT_ASYNCCNT() asm volatile("s_wait_asynccnt 0x0" ::: "memory")
#  endif
#else
#  define USE_TDM 0
#  define USE_ASYNC_LDS 0
#  define WAIT_ASYNCCNT() asm volatile("s_wait_asynccnt 0x0" ::: "memory")
#endif

#define LDS_OFF(p) ((unsigned)(unsigned long long)(l3char*)(p))

#if USE_ASYNC_LDS
#define GLD_ASYNC_B128(gsrc, ldst) \
  __builtin_amdgcn_global_load_async_to_lds_b128((g1void*)(gsrc), (l3void*)(ldst), 0, 0)
#endif

#if USE_TDM
// 2D tile TDM load: tile0 elems per row (contiguous), tile1 rows, row stride
// `stride0` elems, element size 2^dszLog2 bytes. Rows are 128B here, and we
// use the TDM pad feature (pad_interval=32 dwords, pad_amount=4 dwords) to
// produce a 144B LDS pitch (= 36 floats / 72 bf16) for bank-conflict padding.
__device__ __forceinline__ void tdm_load_2d(const void* gsrc, unsigned lds_off,
                                            unsigned tile0, unsigned tile1,
                                            unsigned stride0, unsigned dszLog2)
{
  const unsigned long long ga = (unsigned long long)gsrc;
  v4u g0 = { 1u,                                        // count=1 (valid), no gather
             lds_off,                                   // lds_addr
             (unsigned)(ga & 0xffffffffu),              // global_addr[31:0]
             (unsigned)((ga >> 32) & 0x01ffffffu) | (2u << 30) };  // addr[56:32] | type=2
  const unsigned w0 = (dszLog2 << 16)   // data_size
                    | (1u << 20)        // pad_enable
                    | (4u << 22)        // pad_interval: 32 dwords (one 128B row)
                    | (3u << 25);       // pad_amount: 4 dwords (16B)
  v8i g1 = { (int)w0,
             (int)((tile0 & 0xffffu) << 16),                        // tensor_dim0 lo
             (int)(((tile1 & 0xffffu) << 16) | (tile0 >> 16)),      // tdim1 lo | tdim0 hi
             (int)(((tile0 & 0xffffu) << 16) | (tile1 >> 16)),      // tile_dim0 | tdim1 hi
             (int)(tile1 & 0xffffu),                                // tile_dim1 (tile_dim2=0)
             (int)stride0,                                          // tensor_dim0_stride lo
             0, 0 };
  v4i g2 = { 0, 0, 0, 0 }, g3 = { 0, 0, 0, 0 };
#if defined(__clang_major__) && (__clang_major__ >= 23)
  // therock / clang-23 toolchain: 6-arg form (extra int32x8, zero-filled)
  v8i g4 = { 0, 0, 0, 0, 0, 0, 0, 0 };
  __builtin_amdgcn_tensor_load_to_lds(g0, g1, g2, g3, g4, 0);
#else
  // ROCm 7.2 / clang-22 toolchain: 5-arg form
  __builtin_amdgcn_tensor_load_to_lds(g0, g1, g2, g3, 0);
#endif
}
#endif

// Problem constants: B=4, L=2048, D=1024, H=16, HD=64, M_TOT=8192.

// ---------------------------------------------------------------------------
// GEMM: Y[8192,1024] = A[8192,1024] @ W^T + bias (W is [N,K] row-major).
// WG tile 128x64, 8 waves x (32x32), K-step 32, bf16 hi/lo compensation,
// double-buffered LDS staging via TDM / async-to-LDS.
// MODE 0: -> per-head [BH,L,64] bf16 hi+lo.   MODE 1: -> V^T [BH,64,L] bf16 hi.
// MODE 2: -> fp32 [M,N].
// ---------------------------------------------------------------------------
template <int MODE>
__global__ __launch_bounds__(256)
void gemm128x64(const float* __restrict__ A, const float* __restrict__ W,
                const float* __restrict__ bias,
                bf16* __restrict__ outHi, bf16* __restrict__ outLo,
                float* __restrict__ outF)
{
  __shared__ float sA[2][128 * 36];   // 128 rows x 32 cols (+4 pad)
  __shared__ float sB[2][64 * 36];

  const int tid  = threadIdx.x;
  const int lane = tid & 31;
  const int w    = tid >> 5;
  const int wm   = w & 3;
  const int wn   = w >> 2;
  const int mBlk = blockIdx.x * 128;
  const int nBlk = blockIdx.y * 64;

  const int halfSel = (lane >> 4) & 1;
  const int lr      = lane & 15;
  const int aBase   = halfSel * 16;
  const int bKoff   = halfSel * 16;

  const int ar = tid >> 1, ac = (tid & 1) * 16;
  const int br = tid >> 2, bc = (tid & 3) * 8;

  v8f acc[2][2] = {};

  auto stage = [&](int p, int kk) {
#if USE_TDM
    if (tid < 32) {
      tdm_load_2d(A + (size_t)mBlk * 1024 + kk, LDS_OFF(&sA[p][0]), 32u, 128u, 1024u, 2u);
      tdm_load_2d(W + (size_t)nBlk * 1024 + kk, LDS_OFF(&sB[p][0]), 32u, 64u, 1024u, 2u);
    }
#elif USE_ASYNC_LDS
    {
      const float* srcA = A + (size_t)(mBlk + ar) * 1024 + kk + ac;
      float* dstA = &sA[p][ar * 36 + ac];
      GLD_ASYNC_B128(srcA + 0,  dstA + 0);
      GLD_ASYNC_B128(srcA + 4,  dstA + 4);
      GLD_ASYNC_B128(srcA + 8,  dstA + 8);
      GLD_ASYNC_B128(srcA + 12, dstA + 12);
      const float* srcB = W + (size_t)(nBlk + br) * 1024 + kk + bc;
      float* dstB = &sB[p][br * 36 + bc];
      GLD_ASYNC_B128(srcB + 0, dstB + 0);
      GLD_ASYNC_B128(srcB + 4, dstB + 4);
    }
#else
    {
      const float* src = A + (size_t)(mBlk + ar) * 1024 + kk + ac;
      float* dst = &sA[p][ar * 36 + ac];
      *(float4*)(dst + 0)  = *(const float4*)(src + 0);
      *(float4*)(dst + 4)  = *(const float4*)(src + 4);
      *(float4*)(dst + 8)  = *(const float4*)(src + 8);
      *(float4*)(dst + 12) = *(const float4*)(src + 12);
      const float* srcB = W + (size_t)(nBlk + br) * 1024 + kk + bc;
      float* dstB = &sB[p][br * 36 + bc];
      *(float4*)(dstB + 0) = *(const float4*)(srcB + 0);
      *(float4*)(dstB + 4) = *(const float4*)(srcB + 4);
    }
#endif
  };
  auto wait_stage = [&]() {
#if USE_TDM
    if (tid < 32) __builtin_amdgcn_s_wait_tensorcnt(0);
#elif USE_ASYNC_LDS
    WAIT_ASYNCCNT();
#endif
  };

  stage(0, 0);
  for (int kk = 0; kk < 1024; kk += 32) {
    const int p = (kk >> 5) & 1;
    wait_stage();
    __syncthreads();
    if (kk + 32 < 1024) stage(p ^ 1, kk + 32);

    v16bf ah[2], al[2];
#pragma unroll
    for (int mi = 0; mi < 2; ++mi) {
      const float* row = &sA[p][(wm * 32 + mi * 16 + lr) * 36];
#pragma unroll
      for (int e = 0; e < 8; ++e) {
        float v0 = row[aBase + e];
        float v1 = row[aBase + 8 + e];
        bf16 h0 = (bf16)v0, h1 = (bf16)v1;
        ah[mi][e]     = h0;  al[mi][e]     = (bf16)(v0 - (float)h0);
        ah[mi][e + 8] = h1;  al[mi][e + 8] = (bf16)(v1 - (float)h1);
      }
    }
#pragma unroll
    for (int ni = 0; ni < 2; ++ni) {
      const float* rowB = &sB[p][(wn * 32 + ni * 16 + lr) * 36];
      v16bf bh, bl;
#pragma unroll
      for (int e = 0; e < 16; ++e) {
        float v = rowB[bKoff + e];
        bf16 h = (bf16)v;
        bh[e] = h;  bl[e] = (bf16)(v - (float)h);
      }
#pragma unroll
      for (int mi = 0; mi < 2; ++mi) {
        acc[mi][ni] = WMMA_BF16(ah[mi], bh, acc[mi][ni]);
        acc[mi][ni] = WMMA_BF16(al[mi], bh, acc[mi][ni]);
        acc[mi][ni] = WMMA_BF16(ah[mi], bl, acc[mi][ni]);
      }
    }
    __syncthreads();
  }

#pragma unroll
  for (int mi = 0; mi < 2; ++mi) {
#pragma unroll
    for (int ni = 0; ni < 2; ++ni) {
      const int nG = nBlk + wn * 32 + ni * 16 + lr;
      const float bval = bias[nG];
#pragma unroll
      for (int r = 0; r < 8; ++r) {
        const int mG = mBlk + wm * 32 + mi * 16 + halfSel * 8 + r;
        const float y = acc[mi][ni][r] + bval;
        if constexpr (MODE == 2) {
          outF[(size_t)mG * 1024 + nG] = y;
        } else {
          const int hh = nG >> 6, dd = nG & 63;
          const int bb = mG >> 11, ll = mG & 2047;
          const bf16 hy = (bf16)y;
          if constexpr (MODE == 0) {
            const size_t idx = (((size_t)(bb * 16 + hh)) * 2048 + ll) * 64 + dd;
            outHi[idx] = hy;
            outLo[idx] = (bf16)(y - (float)hy);
          } else {  // MODE 1: V stored transposed per head: [BH, 64, L]
            const size_t idx = (((size_t)(bb * 16 + hh)) * 64 + dd) * 2048 + ll;
            outHi[idx] = hy;
          }
        }
      }
    }
  }
}

// ---------------------------------------------------------------------------
// Flash attention: WG = 64 q-rows of one (b,h); 4 waves x 16 rows.
// Double-buffered K(hi/lo)/V^T tiles staged by TDM / async-to-LDS.
// ---------------------------------------------------------------------------
__global__ __launch_bounds__(128)
void attn64(const bf16* __restrict__ qh, const bf16* __restrict__ ql,
            const bf16* __restrict__ kh, const bf16* __restrict__ kl,
            const bf16* __restrict__ vt, const int* __restrict__ amask,
            float* __restrict__ ctx)
{
  __shared__ bf16 sKh[2][64 * 72];     // [key][d], 16B row pad
  __shared__ bf16 sKl[2][64 * 72];
  __shared__ bf16 sVt[2][64 * 72];     // [d][key]
  __shared__ float sP[4][16 * 68];     // per-wave P transpose buffer

  const int tid  = threadIdx.x;
  const int lane = tid & 31;
  const int w    = tid >> 5;
  const int halfSel = (lane >> 4) & 1;
  const int lr      = lane & 15;
  const int bh = blockIdx.y;
  const int b  = bh >> 4, h = bh & 15;
  const int q0 = blockIdx.x * 64;
  const size_t headOff = (size_t)bh * 2048 * 64;

  // resident Q fragments (hi+lo)
  v16bf aQh[2], aQl[2];
  {
    const int qrow = q0 + w * 16 + lr;
    const bf16* qph = qh + headOff + (size_t)qrow * 64;
    const bf16* qpl = ql + headOff + (size_t)qrow * 64;
#pragma unroll
    for (int ks = 0; ks < 2; ++ks) {
      const int base = ks * 32 + halfSel * 16;
#pragma unroll
      for (int e = 0; e < 8; ++e) {
        aQh[ks][e]     = qph[base + e];
        aQh[ks][e + 8] = qph[base + 8 + e];
        aQl[ks][e]     = qpl[base + e];
        aQl[ks][e + 8] = qpl[base + 8 + e];
      }
    }
  }

  v8f acc[4] = {};
  float mrow[8], lrow[8];
#pragma unroll
  for (int r = 0; r < 8; ++r) { mrow[r] = -3.0e38f; lrow[r] = 0.f; }

  auto stageKV = [&](int p, int j0) {
#if USE_TDM
    if (tid < 32) {
      tdm_load_2d(kh + headOff + (size_t)j0 * 64, LDS_OFF(&sKh[p][0]), 64u, 64u, 64u, 1u);
      tdm_load_2d(kl + headOff + (size_t)j0 * 64, LDS_OFF(&sKl[p][0]), 64u, 64u, 64u, 1u);
      tdm_load_2d(vt + (size_t)bh * 64 * 2048 + j0, LDS_OFF(&sVt[p][0]), 64u, 64u, 2048u, 1u);
    }
#elif USE_ASYNC_LDS
    {
      const int row = tid >> 1, part = (tid & 1) * 32;
      const bf16* s0 = kh + headOff + (size_t)(j0 + row) * 64 + part;
      const bf16* s1 = kl + headOff + (size_t)(j0 + row) * 64 + part;
      const bf16* s2 = vt + ((size_t)bh * 64 + row) * 2048 + j0 + part;
      bf16* d0 = &sKh[p][row * 72 + part];
      bf16* d1 = &sKl[p][row * 72 + part];
      bf16* d2 = &sVt[p][row * 72 + part];
#pragma unroll
      for (int i = 0; i < 4; ++i) {
        GLD_ASYNC_B128(s0 + i * 8, d0 + i * 8);
        GLD_ASYNC_B128(s1 + i * 8, d1 + i * 8);
        GLD_ASYNC_B128(s2 + i * 8, d2 + i * 8);
      }
    }
#else
    {
      const int row = tid >> 1, part = (tid & 1) * 32;
      const uint4* s0 = (const uint4*)(kh + headOff + (size_t)(j0 + row) * 64 + part);
      const uint4* s1 = (const uint4*)(kl + headOff + (size_t)(j0 + row) * 64 + part);
      const uint4* s2 = (const uint4*)(vt + ((size_t)bh * 64 + row) * 2048 + j0 + part);
      uint4* d0 = (uint4*)&sKh[p][row * 72 + part];
      uint4* d1 = (uint4*)&sKl[p][row * 72 + part];
      uint4* d2 = (uint4*)&sVt[p][row * 72 + part];
#pragma unroll
      for (int i = 0; i < 4; ++i) { d0[i] = s0[i]; d1[i] = s1[i]; d2[i] = s2[i]; }
    }
#endif
  };
  auto wait_stage = [&]() {
#if USE_TDM
    if (tid < 32) __builtin_amdgcn_s_wait_tensorcnt(0);
#elif USE_ASYNC_LDS
    WAIT_ASYNCCNT();
#endif
  };

  stageKV(0, 0);
  for (int it = 0; it < 32; ++it) {
    const int j0 = it * 64;
    const int p  = it & 1;
    wait_stage();
    __syncthreads();
    if (it + 1 < 32) stageKV(p ^ 1, j0 + 64);

    // ---- S = (Q K^T) / sqrt(hd) ----
    v8f sc[4];
#pragma unroll
    for (int nt = 0; nt < 4; ++nt) {
      v8f c = {};
      const bf16* krow_h = &sKh[p][(nt * 16 + lr) * 72];
      const bf16* krow_l = &sKl[p][(nt * 16 + lr) * 72];
#pragma unroll
      for (int ks = 0; ks < 2; ++ks) {
        const int koff = ks * 32 + halfSel * 16;
        v16bf bh_, bl_;
#pragma unroll
        for (int e = 0; e < 16; ++e) { bh_[e] = krow_h[koff + e]; bl_[e] = krow_l[koff + e]; }
        c = WMMA_BF16(aQh[ks], bh_, c);
        c = WMMA_BF16(aQl[ks], bh_, c);
        c = WMMA_BF16(aQh[ks], bl_, c);
      }
      sc[nt] = c;
    }

    // ---- mask + scale ----
    float pbuf[4][8];
#pragma unroll
    for (int nt = 0; nt < 4; ++nt) {
      const bool masked = (amask[b * 2048 + j0 + nt * 16 + lr] == 0);
#pragma unroll
      for (int r = 0; r < 8; ++r) {
        const float s = sc[nt][r] * 0.125f;
        pbuf[nt][r] = masked ? -1.0e30f : s;
      }
    }

    // ---- online softmax ----
#pragma unroll
    for (int r = 0; r < 8; ++r) {
      float rmax = fmaxf(fmaxf(pbuf[0][r], pbuf[1][r]), fmaxf(pbuf[2][r], pbuf[3][r]));
#pragma unroll
      for (int off = 8; off >= 1; off >>= 1)
        rmax = fmaxf(rmax, __shfl_xor(rmax, off, 32));
      const float newm = fmaxf(mrow[r], rmax);
      const float corr = __expf(mrow[r] - newm);
      float rsum = 0.f;
#pragma unroll
      for (int nt = 0; nt < 4; ++nt) {
        const float pv = (pbuf[nt][r] <= -1.0e29f) ? 0.f : __expf(pbuf[nt][r] - newm);
        pbuf[nt][r] = pv;
        rsum += pv;
      }
#pragma unroll
      for (int off = 8; off >= 1; off >>= 1)
        rsum += __shfl_xor(rsum, off, 32);
      lrow[r] = lrow[r] * corr + rsum;
      mrow[r] = newm;
#pragma unroll
      for (int nt = 0; nt < 4; ++nt) acc[nt][r] *= corr;
    }

    // ---- transpose P through LDS (C layout -> A layout) ----
#pragma unroll
    for (int nt = 0; nt < 4; ++nt)
#pragma unroll
      for (int r = 0; r < 8; ++r)
        sP[w][(halfSel * 8 + r) * 68 + nt * 16 + lr] = pbuf[nt][r];
    __syncthreads();

    // ---- ctx += P * V ----
#pragma unroll
    for (int ks = 0; ks < 2; ++ks) {
      v16bf ap;
      const float* prow = &sP[w][lr * 68];
      const int base = ks * 32 + halfSel * 16;
#pragma unroll
      for (int e = 0; e < 8; ++e) {
        ap[e]     = (bf16)prow[base + e];
        ap[e + 8] = (bf16)prow[base + 8 + e];
      }
      const int koff = ks * 32 + halfSel * 16;
#pragma unroll
      for (int nt = 0; nt < 4; ++nt) {
        const bf16* vrow = &sVt[p][(nt * 16 + lr) * 72];
        v16bf bv;
#pragma unroll
        for (int e = 0; e < 16; ++e) bv[e] = vrow[koff + e];
        acc[nt] = WMMA_BF16(ap, bv, acc[nt]);
      }
    }
  }

  // ---- epilogue ----
#pragma unroll
  for (int r = 0; r < 8; ++r) {
    const int qrow = q0 + w * 16 + halfSel * 8 + r;
    const float qm  = (float)amask[b * 2048 + qrow];
    const float inv = (lrow[r] > 0.f) ? (qm / lrow[r]) : 0.f;
#pragma unroll
    for (int nt = 0; nt < 4; ++nt) {
      const int col = h * 64 + nt * 16 + lr;
      ctx[((size_t)(b * 2048 + qrow)) * 1024 + col] = acc[nt][r] * inv;
    }
  }
}

// ---------------------------------------------------------------------------
extern "C" void kernel_launch(void* const* d_in, const int* in_sizes, int n_in,
                              void* d_out, int out_size, void* d_ws, size_t ws_size,
                              hipStream_t stream)
{
  const float* x   = (const float*)d_in[0];
  const int*   am  = (const int*)d_in[1];
  const float* wq  = (const float*)d_in[2];
  const float* bq  = (const float*)d_in[3];
  const float* wk  = (const float*)d_in[4];
  const float* bk  = (const float*)d_in[5];
  const float* wv  = (const float*)d_in[6];
  const float* bv  = (const float*)d_in[7];
  const float* wo  = (const float*)d_in[8];
  const float* bo  = (const float*)d_in[9];
  float* out = (float*)d_out;

  char* ws = (char*)d_ws;
  const size_t MB = 1024u * 1024u;
  bf16*  qh_ = (bf16*)(ws + 0 * MB);    // [BH, L, 64] bf16 hi
  bf16*  ql_ = (bf16*)(ws + 16 * MB);   // lo
  bf16*  kh_ = (bf16*)(ws + 32 * MB);
  bf16*  kl_ = (bf16*)(ws + 48 * MB);
  bf16*  vt_ = (bf16*)(ws + 64 * MB);   // V^T per head: [BH, 64, L]
  float* ctx = (float*)(ws + 80 * MB);  // [B, L, D] fp32

  dim3 gGrid(64, 16), gBlk(256);
  gemm128x64<0><<<gGrid, gBlk, 0, stream>>>(x, wq, bq, qh_, ql_, nullptr);
  gemm128x64<0><<<gGrid, gBlk, 0, stream>>>(x, wk, bk, kh_, kl_, nullptr);
  gemm128x64<1><<<gGrid, gBlk, 0, stream>>>(x, wv, bv, vt_, nullptr, nullptr);

  attn64<<<dim3(32, 64), dim3(128), 0, stream>>>(qh_, ql_, kh_, kl_, vt_, am, ctx);

  gemm128x64<2><<<gGrid, gBlk, 0, stream>>>(ctx, wo, bo, nullptr, nullptr, out);
}
